// HierarchicalWTABlockV2_27144193310736
// MI455X (gfx1250) — matrix operand order repaired and checked by reading, not compiled, not run on hardware
//
#include <hip/hip_runtime.h>
#include <hip/hip_bf16.h>
#include <math.h>

// ---------------------------------------------------------------------------
// Types / fragment helpers for v_wmma_f32_16x16x32_bf16 (wave32, gfx1250)
// ---------------------------------------------------------------------------
typedef __attribute__((ext_vector_type(16))) __bf16 v16bf;
typedef __attribute__((ext_vector_type(8)))  float  v8f;

union Frag {
    v16bf v;
    unsigned short u[16];
    unsigned int   d[8];
};

#define DEV __device__ __forceinline__

DEV unsigned short f2bf(float f) {
    unsigned int x = __float_as_uint(f);
    unsigned int r = x + 0x7FFFu + ((x >> 16) & 1u);   // RNE
    return (unsigned short)(r >> 16);
}
DEV float bf2f(unsigned short h) { return __uint_as_float(((unsigned int)h) << 16); }

DEV float gelu_exact(float x) { return 0.5f * x * (1.0f + erff(x * 0.70710678118654752f)); }

DEV v8f zero8() {
    v8f c = {0.f, 0.f, 0.f, 0.f, 0.f, 0.f, 0.f, 0.f};
    return c;
}

DEV v8f wmma_bf16(const Frag& a, const Frag& b, v8f c) {
    return __builtin_amdgcn_wmma_f32_16x16x32_bf16(false, a.v, false, b.v,
                                                   (short)0, c, false, false);
}

// A fragment (16x32 bf16, row-major source, base pre-offset to tile origin).
// Lane L<16: row=L, K 0..15 ; Lane L>=16: row=L-16, K 16..31.
DEV void loadA(Frag& f, const unsigned short* base, int lda, int lane) {
    int m  = lane & 15;
    int kb = (lane >> 4) << 4;
    const unsigned short* row = base + m * lda + kb;
#pragma unroll
    for (int r = 0; r < 8; ++r) {
        int kk = 2 * r;
        f.d[r] = *(const unsigned int*)(row + kk);
    }
}

// B fragment (32x16 bf16) from a TRANSPOSED weight (W^T row-major, N x K),
// base pre-offset to (n0,k0). Lane L<16: K 0..15 ; L>=16: K 16..31.
DEV void loadBT(Frag& f, const unsigned short* base, int ldk, int lane) {
    int n  = lane & 15;
    int kb = (lane >> 4) << 4;
    const unsigned short* row = base + n * ldk + kb;
#pragma unroll
    for (int r = 0; r < 8; ++r) f.d[r] = *(const unsigned int*)(row + 2 * r);
}

// B fragment (32x16 bf16) from a row-major K x N matrix, base pre-offset.
DEV void loadB_rm(Frag& f, const unsigned short* base, int ldb, int lane) {
    int n  = lane & 15;
    int kb = (lane >> 4) << 4;
    const unsigned short* p = base + kb * ldb + n;
#pragma unroll
    for (int r = 0; r < 8; ++r) {
        f.u[2 * r]     = p[(2 * r) * ldb];
        f.u[2 * r + 1] = p[(2 * r + 1) * ldb];
    }
}

// ---------------------------------------------------------------------------
// Problem constants
// ---------------------------------------------------------------------------
#define BB 8
#define LL 4096
#define DD 256
#define GG 64
#define KK 8
#define NN 512
#define HH 8
#define DH 32

// ---------------------------------------------------------------------------
// Weight transpose-cast fp32 -> bf16 : dst[c*rows + r] = src[r*cols + c]
// ---------------------------------------------------------------------------
__global__ void k_castT(const float* __restrict__ src, unsigned short* __restrict__ dst,
                        int rows, int cols) {
    int i = blockIdx.x * blockDim.x + threadIdx.x;
    if (i < rows * cols) {
        int r = i / cols, c = i % cols;
        dst[(size_t)c * rows + r] = f2bf(src[i]);
    }
}

// ---------------------------------------------------------------------------
// Token path: msg/gate MLPs (WMMA bf16), fp32 routing + argmax, atomic scatter
// One block = 32 tokens, 256 threads (8 waves). B fragments are reused in
// registers across the two 16-row tiles (2 wmma per B load).
// ---------------------------------------------------------------------------
__global__ __launch_bounds__(256) void k_token(
    const float* __restrict__ X, const float* __restrict__ Wg, const float* __restrict__ Wk,
    const unsigned short* __restrict__ Wm1T, const float* __restrict__ bm1,
    const unsigned short* __restrict__ Wm2T, const float* __restrict__ bm2,
    const unsigned short* __restrict__ Wgt1T, const float* __restrict__ bgt1,
    const float* __restrict__ Wgt2, const float* __restrict__ bgt2,
    float* __restrict__ incoming) {
    __shared__ __align__(16) unsigned short Xs[32 * DD];   // X tile bf16
    __shared__ __align__(16) unsigned short Hs[32 * 512];  // h1, then g1
    __shared__ float lgG[32 * GG];
    __shared__ float lgK[32 * KK];
    __shared__ float gates[32];
    __shared__ int   idxs[32];

    const int tid = threadIdx.x, lane = tid & 31, wave = tid >> 5;
    const int rowBase = blockIdx.x * 32;      // flat token index in [0, B*L)
    const int b = rowBase >> 12;              // / L

    // stage X tile as bf16
    for (int i = tid; i < 32 * DD; i += 256) Xs[i] = f2bf(X[(size_t)rowBase * DD + i]);

    // fp32 routing logits (exact path for argmax fidelity)
    for (int i = tid; i < 32 * GG; i += 256) {
        int r = i >> 6, g = i & 63;
        const float* xr = X + (size_t)(rowBase + r) * DD;
        float acc = 0.f;
        for (int d = 0; d < DD; ++d) acc += xr[d] * Wg[d * GG + g];
        lgG[i] = acc;
    }
    for (int i = tid; i < 32 * KK; i += 256) {
        int r = i >> 3, g = i & 7;
        const float* xr = X + (size_t)(rowBase + r) * DD;
        float acc = 0.f;
        for (int d = 0; d < DD; ++d) acc += xr[d] * Wk[d * KK + g];
        lgK[i] = acc;
    }
    __syncthreads();

    if (tid < 32) {   // first-occurrence argmax (matches jnp.argmax)
        float bg = lgG[tid * GG]; int gi = 0;
        for (int j = 1; j < GG; ++j) { float v = lgG[tid * GG + j]; if (v > bg) { bg = v; gi = j; } }
        float bk = lgK[tid * KK]; int ki = 0;
        for (int j = 1; j < KK; ++j) { float v = lgK[tid * KK + j]; if (v > bk) { bk = v; ki = j; } }
        idxs[tid] = gi * KK + ki;
    }
    __syncthreads();

    Frag a, bb;
    const int n = lane & 15, hi = lane >> 4;

    // h1 = gelu(X @ Wm1 + bm1) : 32x512 ; tn per wave, both row tiles share B
#pragma unroll
    for (int j = 0; j < 4; ++j) {
        int tn = wave + 8 * j;
        if (j < 3) __builtin_prefetch(&Wm1T[(size_t)((tn + 8) * 16) * DD], 0, 0);
        v8f c0 = zero8(), c1 = zero8();
#pragma unroll
        for (int kt = 0; kt < 8; ++kt) {
            loadBT(bb, &Wm1T[(tn * 16) * DD + kt * 32], DD, lane);
            loadA(a, &Xs[kt * 32], DD, lane);
            c0 = wmma_bf16(a, bb, c0);
            loadA(a, &Xs[16 * DD + kt * 32], DD, lane);
            c1 = wmma_bf16(a, bb, c1);
        }
        float bv = bm1[tn * 16 + n];
#pragma unroll
        for (int i = 0; i < 8; ++i) {
            int m = i + hi * 8;
            Hs[m * 512 + tn * 16 + n]        = f2bf(gelu_exact(c0[i] + bv));
            Hs[(16 + m) * 512 + tn * 16 + n] = f2bf(gelu_exact(c1[i] + bv));
        }
    }
    __syncthreads();

    // msg = h1 @ Wm2 + bm2 : 32x256, kept in registers (4 tiles / wave)
    v8f cm[4];
#pragma unroll
    for (int j = 0; j < 2; ++j) {
        int tn = wave + 8 * j;
        v8f c0 = zero8(), c1 = zero8();
#pragma unroll
        for (int kt = 0; kt < 16; ++kt) {
            loadBT(bb, &Wm2T[(tn * 16) * 512 + kt * 32], 512, lane);
            loadA(a, &Hs[kt * 32], 512, lane);
            c0 = wmma_bf16(a, bb, c0);
            loadA(a, &Hs[16 * 512 + kt * 32], 512, lane);
            c1 = wmma_bf16(a, bb, c1);
        }
        float bv = bm2[tn * 16 + n];
#pragma unroll
        for (int i = 0; i < 8; ++i) { c0[i] += bv; c1[i] += bv; }
        cm[2 * j] = c0; cm[2 * j + 1] = c1;
    }
    __syncthreads();

    // g1 = gelu(X @ Wgt1 + bgt1) : 32x256 (reuse Hs, ld=256)
#pragma unroll
    for (int j = 0; j < 2; ++j) {
        int tn = wave + 8 * j;
        v8f c0 = zero8(), c1 = zero8();
#pragma unroll
        for (int kt = 0; kt < 8; ++kt) {
            loadBT(bb, &Wgt1T[(tn * 16) * DD + kt * 32], DD, lane);
            loadA(a, &Xs[kt * 32], DD, lane);
            c0 = wmma_bf16(a, bb, c0);
            loadA(a, &Xs[16 * DD + kt * 32], DD, lane);
            c1 = wmma_bf16(a, bb, c1);
        }
        float bv = bgt1[tn * 16 + n];
#pragma unroll
        for (int i = 0; i < 8; ++i) {
            int m = i + hi * 8;
            Hs[m * 256 + tn * 16 + n]        = f2bf(gelu_exact(c0[i] + bv));
            Hs[(16 + m) * 256 + tn * 16 + n] = f2bf(gelu_exact(c1[i] + bv));
        }
    }
    __syncthreads();

    // gate = sigmoid(g1 . Wgt2 + bgt2) ; 8 threads per row + shfl reduce
    {
        int r = tid >> 3, j = tid & 7;
        float acc = 0.f;
        for (int d = j * 32; d < j * 32 + 32; ++d) acc += bf2f(Hs[r * 256 + d]) * Wgt2[d];
#pragma unroll
        for (int w = 1; w < 8; w <<= 1) acc += __shfl_xor(acc, w, 32);
        if (j == 0) gates[r] = 1.f / (1.f + expf(-(acc + bgt2[0])));
    }
    __syncthreads();

    // scatter gate*msg into incoming[b, slot, :]
#pragma unroll
    for (int j = 0; j < 2; ++j) {
        int tn = wave + 8 * j;
#pragma unroll
        for (int tm = 0; tm < 2; ++tm) {
            v8f c = cm[2 * j + tm];
#pragma unroll
            for (int i = 0; i < 8; ++i) {
                int m = tm * 16 + i + hi * 8;
                float val = c[i] * gates[m];
                int slot = idxs[m];
                atomicAdd(&incoming[((size_t)b * NN + slot) * DD + tn * 16 + n], val);
            }
        }
    }
}

// ---------------------------------------------------------------------------
// QKV projection: S @ Wqkv + bqkv -> q[bh][n][dh], kT[bh][dh][n], v[bh][n][dh]
// One block = 32 slots, 256 threads; B fragment shared across both row tiles.
// ---------------------------------------------------------------------------
__global__ __launch_bounds__(256) void k_qkv(
    const float* __restrict__ S, const unsigned short* __restrict__ WqkvT,
    const float* __restrict__ bqkv,
    unsigned short* __restrict__ q, unsigned short* __restrict__ kT,
    unsigned short* __restrict__ v) {
    __shared__ __align__(16) unsigned short Ss[32 * DD];
    const int tid = threadIdx.x, lane = tid & 31, wave = tid >> 5;
    const int rowBase = blockIdx.x * 32;          // in [0, B*N)
    const int b = rowBase >> 9;

    for (int i = tid; i < 32 * DD; i += 256) Ss[i] = f2bf(S[(size_t)rowBase * DD + i]);
    __syncthreads();

    Frag a, bb;
    const int n = lane & 15, hi = lane >> 4;
#pragma unroll
    for (int j = 0; j < 6; ++j) {                  // 48 col tiles / 8 waves
        int tn = wave + 8 * j;
        if (j < 5) __builtin_prefetch(&WqkvT[(size_t)((tn + 8) * 16) * DD], 0, 0);
        v8f c0 = zero8(), c1 = zero8();
#pragma unroll
        for (int kt = 0; kt < 8; ++kt) {
            loadBT(bb, &WqkvT[(tn * 16) * DD + kt * 32], DD, lane);
            loadA(a, &Ss[kt * 32], DD, lane);
            c0 = wmma_bf16(a, bb, c0);
            loadA(a, &Ss[16 * DD + kt * 32], DD, lane);
            c1 = wmma_bf16(a, bb, c1);
        }
        int col = tn * 16 + n;
        float bv = bqkv[col];
#pragma unroll
        for (int tm = 0; tm < 2; ++tm) {
            v8f c = tm ? c1 : c0;
#pragma unroll
            for (int i = 0; i < 8; ++i) {
                int slot = (rowBase + tm * 16 + i + hi * 8) & (NN - 1);
                unsigned short bvv = f2bf(c[i] + bv);
                if (col < 256) {
                    int h = col >> 5, dh = col & 31;
                    q[(((size_t)b * HH + h) * NN + slot) * DH + dh] = bvv;
                } else if (col < 512) {
                    int cc = col - 256, h = cc >> 5, dh = cc & 31;
                    kT[(((size_t)b * HH + h) * DH + dh) * NN + slot] = bvv;
                } else {
                    int cc = col - 512, h = cc >> 5, dh = cc & 31;
                    v[(((size_t)b * HH + h) * NN + slot) * DH + dh] = bvv;
                }
            }
        }
    }
}

// ---------------------------------------------------------------------------
// Attention with online softmax. Block = one (b,h), 128 queries, 8 waves.
// Each wave: 16 queries; key tiles of 32 keys; per tile: 2 score WMMA + 2 PV WMMA.
// ---------------------------------------------------------------------------
__global__ __launch_bounds__(256) void k_attn(
    const unsigned short* __restrict__ q, const unsigned short* __restrict__ kT,
    const unsigned short* __restrict__ v, unsigned short* __restrict__ o) {
    __shared__ __align__(16) unsigned short Ps[8][16 * 32];
    const int lane = threadIdx.x & 31, wave = threadIdx.x >> 5;
    const int bh = blockIdx.x >> 2;               // b*H + h
    const int q0 = (blockIdx.x & 3) * 128 + wave * 16;
    const unsigned short* qb = q  + (size_t)bh * NN * DH;
    const unsigned short* kb = kT + (size_t)bh * DH * NN;
    const unsigned short* vb = v  + (size_t)bh * NN * DH;

    Frag aq, ap, bk0, bk1, bv0, bv1;
    loadA(aq, qb + (size_t)q0 * DH, DH, lane);    // 16x32 query fragment

    v8f o0 = zero8(), o1 = zero8();
    float mrow[8], lrow[8];
#pragma unroll
    for (int i = 0; i < 8; ++i) { mrow[i] = -3.0e38f; lrow[i] = 0.f; }
    const float scale = 0.17677669529663687f;     // 1/sqrt(32)
    const int n = lane & 15, hi = lane >> 4;
    unsigned short* pw = &Ps[wave][0];

    for (int kt = 0; kt < 16; ++kt) {
        const int kbase = kt * 32;
        loadB_rm(bk0, kb + kbase, NN, lane);
        loadB_rm(bk1, kb + kbase + 16, NN, lane);
        v8f s0 = zero8(), s1 = zero8();
        s0 = wmma_bf16(aq, bk0, s0);
        s1 = wmma_bf16(aq, bk1, s1);

        float tmx[8];
#pragma unroll
        for (int i = 0; i < 8; ++i) {
            s0[i] *= scale; s1[i] *= scale;
            tmx[i] = fmaxf(s0[i], s1[i]);
        }
        for (int w = 1; w < 16; w <<= 1)
#pragma unroll
            for (int i = 0; i < 8; ++i) tmx[i] = fmaxf(tmx[i], __shfl_xor(tmx[i], w, 32));

        float corr[8], ps[8];
#pragma unroll
        for (int i = 0; i < 8; ++i) {
            float nm = fmaxf(mrow[i], tmx[i]);
            corr[i] = expf(mrow[i] - nm);
            mrow[i] = nm;
            s0[i] = expf(s0[i] - nm);
            s1[i] = expf(s1[i] - nm);
            ps[i] = s0[i] + s1[i];
        }
        for (int w = 1; w < 16; w <<= 1)
#pragma unroll
            for (int i = 0; i < 8; ++i) ps[i] += __shfl_xor(ps[i], w, 32);
#pragma unroll
        for (int i = 0; i < 8; ++i) {
            lrow[i] = lrow[i] * corr[i] + ps[i];
            o0[i] *= corr[i];
            o1[i] *= corr[i];
        }

        // C-layout -> A-layout via per-wave LDS staging (bf16)
#pragma unroll
        for (int i = 0; i < 8; ++i) {
            int m = i + hi * 8;
            pw[m * 32 + n]      = f2bf(s0[i]);
            pw[m * 32 + 16 + n] = f2bf(s1[i]);
        }
        __syncthreads();
        loadA(ap, pw, 32, lane);
        loadB_rm(bv0, vb + (size_t)kbase * DH, DH, lane);
        loadB_rm(bv1, vb + (size_t)kbase * DH + 16, DH, lane);
        o0 = wmma_bf16(ap, bv0, o0);
        o1 = wmma_bf16(ap, bv1, o1);
        __syncthreads();
    }

    const int bI = bh >> 3, h = bh & 7;
#pragma unroll
    for (int i = 0; i < 8; ++i) {
        int m = q0 + i + hi * 8;
        float inv = 1.f / lrow[i];
        size_t base = ((size_t)bI * NN + m) * DD + h * DH;
        o[base + n]      = f2bf(o0[i] * inv);
        o[base + 16 + n] = f2bf(o1[i] * inv);
    }
}

// ---------------------------------------------------------------------------
// Slot update: o@Wo + S -> LN -> concat(incoming) -> Wu1/gelu/Wu2 -> +res -> LN
// One block = 16 slots, 256 threads. LNs parallelized 16 threads/row.
// ---------------------------------------------------------------------------
__global__ __launch_bounds__(256) void k_slot(
    const unsigned short* __restrict__ o, const float* __restrict__ S,
    const unsigned short* __restrict__ WoT, const float* __restrict__ bo,
    const float* __restrict__ aln_g, const float* __restrict__ aln_b,
    const float* __restrict__ incoming,
    const unsigned short* __restrict__ Wu1T, const float* __restrict__ bu1,
    const unsigned short* __restrict__ Wu2T, const float* __restrict__ bu2,
    const float* __restrict__ ln_g, const float* __restrict__ ln_b,
    float* __restrict__ out) {
    __shared__ __align__(16) unsigned short cats[16 * 512];  // [S1 | incoming] bf16
    __shared__ __align__(16) unsigned short us[16 * 512];    // gelu(cat@Wu1) bf16
    __shared__ float S1f[16 * 256];
    const int tid = threadIdx.x, lane = tid & 31, wave = tid >> 5;
    const int rowBase = blockIdx.x * 16;          // in [0, B*N)
    Frag a, bb;
    const int n = lane & 15, hi = lane >> 4;
    const int lr = tid >> 4, lj = tid & 15;       // LN: row, 16-lane group index

    // Phase A: S1pre = o @ Wo + bo + S
#pragma unroll
    for (int s = 0; s < 2; ++s) {
        int t = wave + 8 * s;                      // 16 col tiles
        v8f c = zero8();
#pragma unroll
        for (int kt = 0; kt < 8; ++kt) {
            loadA(a, o + (size_t)rowBase * DD + kt * 32, DD, lane);
            loadBT(bb, &WoT[(t * 16) * DD + kt * 32], DD, lane);
            c = wmma_bf16(a, bb, c);
        }
        float bv = bo[t * 16 + n];
#pragma unroll
        for (int i = 0; i < 8; ++i) {
            int m = i + hi * 8, col = t * 16 + n;
            S1f[m * 256 + col] = c[i] + bv + S[((size_t)rowBase + m) * DD + col];
        }
    }
    __syncthreads();

    // LN (attn): 16 threads per row, shfl-xor tree reduce (wave32-safe groups)
    {
        float part = 0.f;
        for (int c2 = lj * 16; c2 < lj * 16 + 16; ++c2) part += S1f[lr * 256 + c2];
#pragma unroll
        for (int w = 1; w < 16; w <<= 1) part += __shfl_xor(part, w, 32);
        float mu = part * (1.f / DD);
        float vp = 0.f;
        for (int c2 = lj * 16; c2 < lj * 16 + 16; ++c2) {
            float d = S1f[lr * 256 + c2] - mu; vp += d * d;
        }
#pragma unroll
        for (int w = 1; w < 16; w <<= 1) vp += __shfl_xor(vp, w, 32);
        float is = rsqrtf(vp * (1.f / DD) + 1e-5f);
        for (int c2 = lj * 16; c2 < lj * 16 + 16; ++c2) {
            float vN = (S1f[lr * 256 + c2] - mu) * is * aln_g[c2] + aln_b[c2];
            S1f[lr * 256 + c2] = vN;
            cats[lr * 512 + c2] = f2bf(vN);
        }
    }
    for (int i = tid; i < 16 * DD; i += 256) {     // cats high half = incoming
        int r = i >> 8, cI = i & 255;
        cats[r * 512 + 256 + cI] = f2bf(incoming[((size_t)rowBase + r) * DD + cI]);
    }
    __syncthreads();

    // Phase B: U = gelu(cat @ Wu1 + bu1) : 16x512
#pragma unroll
    for (int s = 0; s < 4; ++s) {
        int t = wave + 8 * s;                      // 32 col tiles
        v8f c = zero8();
#pragma unroll
        for (int kt = 0; kt < 16; ++kt) {
            loadA(a, &cats[kt * 32], 512, lane);
            loadBT(bb, &Wu1T[(t * 16) * 512 + kt * 32], 512, lane);
            c = wmma_bf16(a, bb, c);
        }
        float bv = bu1[t * 16 + n];
#pragma unroll
        for (int i = 0; i < 8; ++i)
            us[(i + hi * 8) * 512 + t * 16 + n] = f2bf(gelu_exact(c[i] + bv));
    }
    __syncthreads();

    // Phase C: S2 = S1 + U @ Wu2 + bu2 (accumulate into S1f)
#pragma unroll
    for (int s = 0; s < 2; ++s) {
        int t = wave + 8 * s;
        v8f c = zero8();
#pragma unroll
        for (int kt = 0; kt < 16; ++kt) {
            loadA(a, &us[kt * 32], 512, lane);
            loadBT(bb, &Wu2T[(t * 16) * 512 + kt * 32], 512, lane);
            c = wmma_bf16(a, bb, c);
        }
        float bv = bu2[t * 16 + n];
#pragma unroll
        for (int i = 0; i < 8; ++i) {
            int m = i + hi * 8;
            S1f[m * 256 + t * 16 + n] += c[i] + bv;
        }
    }
    __syncthreads();

    // Final LN -> out (fp32), parallelized
    {
        float part = 0.f;
        for (int c2 = lj * 16; c2 < lj * 16 + 16; ++c2) part += S1f[lr * 256 + c2];
#pragma unroll
        for (int w = 1; w < 16; w <<= 1) part += __shfl_xor(part, w, 32);
        float mu = part * (1.f / DD);
        float vp = 0.f;
        for (int c2 = lj * 16; c2 < lj * 16 + 16; ++c2) {
            float d = S1f[lr * 256 + c2] - mu; vp += d * d;
        }
#pragma unroll
        for (int w = 1; w < 16; w <<= 1) vp += __shfl_xor(vp, w, 32);
        float is = rsqrtf(vp * (1.f / DD) + 1e-5f);
        for (int c2 = lj * 16; c2 < lj * 16 + 16; ++c2)
            out[((size_t)rowBase + lr) * DD + c2] =
                (S1f[lr * 256 + c2] - mu) * is * ln_g[c2] + ln_b[c2];
    }
}

// ---------------------------------------------------------------------------
// Launch
// ---------------------------------------------------------------------------
extern "C" void kernel_launch(void* const* d_in, const int* in_sizes, int n_in,
                              void* d_out, int out_size, void* d_ws, size_t ws_size,
                              hipStream_t stream) {
    (void)in_sizes; (void)n_in; (void)out_size; (void)ws_size;
    const float* X     = (const float*)d_in[0];
    const float* S     = (const float*)d_in[1];
    const float* Wg    = (const float*)d_in[2];
    const float* Wk    = (const float*)d_in[3];
    const float* Wm1   = (const float*)d_in[4];
    const float* bm1   = (const float*)d_in[5];
    const float* Wm2   = (const float*)d_in[6];
    const float* bm2   = (const float*)d_in[7];
    const float* Wgt1  = (const float*)d_in[8];
    const float* bgt1  = (const float*)d_in[9];
    const float* Wgt2  = (const float*)d_in[10];
    const float* bgt2  = (const float*)d_in[11];
    const float* Wqkv  = (const float*)d_in[12];
    const float* bqkv  = (const float*)d_in[13];
    const float* Wo    = (const float*)d_in[14];
    const float* bo    = (const float*)d_in[15];
    const float* aln_g = (const float*)d_in[16];
    const float* aln_b = (const float*)d_in[17];
    const float* Wu1   = (const float*)d_in[18];
    const float* bu1   = (const float*)d_in[19];
    const float* Wu2   = (const float*)d_in[20];
    const float* bu2   = (const float*)d_in[21];
    const float* ln_g  = (const float*)d_in[22];
    const float* ln_b  = (const float*)d_in[23];

    unsigned char* ws = (unsigned char*)d_ws;
    size_t off = 0;
    auto alloc = [&](size_t bytes) -> void* {
        void* p = ws + off;
        off = (off + bytes + 255) & ~(size_t)255;
        return p;
    };
    unsigned short* wm1t  = (unsigned short*)alloc((size_t)512 * 256 * 2);
    unsigned short* wm2t  = (unsigned short*)alloc((size_t)256 * 512 * 2);
    unsigned short* wgt1t = (unsigned short*)alloc((size_t)256 * 256 * 2);
    unsigned short* wqkvt = (unsigned short*)alloc((size_t)768 * 256 * 2);
    unsigned short* wot   = (unsigned short*)alloc((size_t)256 * 256 * 2);
    unsigned short* wu1t  = (unsigned short*)alloc((size_t)512 * 512 * 2);
    unsigned short* wu2t  = (unsigned short*)alloc((size_t)256 * 512 * 2);
    float*          inc   = (float*)alloc((size_t)BB * NN * DD * 4);
    unsigned short* qws   = (unsigned short*)alloc((size_t)BB * NN * DD * 2);
    unsigned short* kws   = (unsigned short*)alloc((size_t)BB * NN * DD * 2);
    unsigned short* vws   = (unsigned short*)alloc((size_t)BB * NN * DD * 2);
    unsigned short* ows   = (unsigned short*)alloc((size_t)BB * NN * DD * 2);

    auto castT = [&](const float* src, unsigned short* dst, int rows, int cols) {
        int nElems = rows * cols;
        k_castT<<<(nElems + 255) / 256, 256, 0, stream>>>(src, dst, rows, cols);
    };
    castT(Wm1, wm1t, 256, 512);
    castT(Wm2, wm2t, 512, 256);
    castT(Wgt1, wgt1t, 256, 256);
    castT(Wqkv, wqkvt, 256, 768);
    castT(Wo, wot, 256, 256);
    castT(Wu1, wu1t, 512, 512);
    castT(Wu2, wu2t, 512, 256);

    hipMemsetAsync(inc, 0, (size_t)BB * NN * DD * 4, stream);

    k_token<<<(BB * LL) / 32, 256, 0, stream>>>(X, Wg, Wk, wm1t, bm1, wm2t, bm2,
                                                wgt1t, bgt1, Wgt2, bgt2, inc);
    k_qkv<<<(BB * NN) / 32, 256, 0, stream>>>(S, wqkvt, bqkv, qws, kws, vws);
    k_attn<<<BB * HH * 4, 256, 0, stream>>>(qws, kws, vws, ows);
    k_slot<<<(BB * NN) / 16, 256, 0, stream>>>(ows, S, wot, bo, aln_g, aln_b, inc,
                                               wu1t, bu1, wu2t, bu2, ln_g, ln_b,
                                               (float*)d_out);
}
